// TTransformer_21474836480553
// MI455X (gfx1250) — matrix operand order repaired and checked by reading, not compiled, not run on hardware
//
#include <hip/hip_runtime.h>
#include <hip/hip_bf16.h>

typedef __attribute__((ext_vector_type(16))) _Float16 v16h;
typedef __attribute__((ext_vector_type(8)))  _Float16 v8h;
typedef __attribute__((ext_vector_type(8)))  float    v8f;

#define NB   4096
#define T    48
#define EE   64
#define TNUM 288
#define LNEPS 1e-5f

// ---- LDS arena (phase-aliased) -------------------------------------------
#define G1_OFF 0
#define G2_OFF 26112
#define G4_OFF 46080
#define G5_OFF 62464
#define SMEM_BYTES 78848

// ---- CDNA5 WMMA helpers (ISA 7.12.2 layouts, wave32) ----------------------
__device__ __forceinline__ v8f wmma16(v16h a, v16h b, v8f c) {
  return __builtin_amdgcn_wmma_f32_16x16x32_f16(false, a, false, b, (short)0, c,
                                                false, false);
}

#if __has_builtin(__builtin_amdgcn_wmma_f16_16x16x32_f16)
typedef v8h acch;
__device__ __forceinline__ acch acch_zero() { return (v8h){}; }
__device__ __forceinline__ acch wmma16h(v16h a, v16h b, acch c) {
  return __builtin_amdgcn_wmma_f16_16x16x32_f16(false, a, false, b, (short)0, c,
                                                false, false);
}
__device__ __forceinline__ _Float16 acch_get(acch c, int i) { return c[i]; }
#else
typedef v8f acch;
__device__ __forceinline__ acch acch_zero() { return (v8f){}; }
__device__ __forceinline__ acch wmma16h(v16h a, v16h b, acch c) {
  return __builtin_amdgcn_wmma_f32_16x16x32_f16(false, a, false, b, (short)0, c,
                                                false, false);
}
__device__ __forceinline__ _Float16 acch_get(acch c, int i) { return (_Float16)c[i]; }
#endif

// A fragment 16x32 f16 from row-major LDS: two aligned b128 loads per lane.
__device__ __forceinline__ v16h ldsA(const _Float16* base, int ld) {
  const int lane = threadIdx.x & 31;
  const int row  = lane & 15;
  const int koff = (lane & 16) ? 16 : 0;
  const _Float16* p = base + row * ld + koff;
  union { v16h v; struct { v8h lo, hi; } s; } u;
  u.s.lo = *(const v8h*)(p);
  u.s.hi = *(const v8h*)(p + 8);
  return u.v;
}

// B fragment 32x16 f16 from paired-K LDS layout (8 aligned b32 loads).
__device__ __forceinline__ v16h ldsBp(const unsigned int* base, int ldw) {
  const int lane = threadIdx.x & 31;
  const int col  = lane & 15;
  const int krow = (lane & 16) ? 8 : 0;
  union { v16h v; unsigned int w[8]; } u;
#pragma unroll
  for (int i = 0; i < 8; ++i) u.w[i] = base[(krow + i) * ldw + col];
  return u.v;
}

__device__ __forceinline__ unsigned int packh2(float a, float b) {
  union { unsigned int u; _Float16 h[2]; } p;
  p.h[0] = (_Float16)a; p.h[1] = (_Float16)b;
  return p.u;
}

__global__ __launch_bounds__(128)
void ttransformer_fused(const float* __restrict__ value,
                        const float* __restrict__ keys,
                        const float* __restrict__ query,
                        const float* __restrict__ Wq, const float* __restrict__ Wk,
                        const float* __restrict__ Wv, const float* __restrict__ Wo,
                        const float* __restrict__ bo, const float* __restrict__ temb,
                        const float* __restrict__ ln1g, const float* __restrict__ ln1b,
                        const float* __restrict__ ln2g, const float* __restrict__ ln2b,
                        const float* __restrict__ W1, const float* __restrict__ b1,
                        const float* __restrict__ W2, const float* __restrict__ b2,
                        const int* __restrict__ tqp, float* __restrict__ out) {
  __shared__ __align__(16) unsigned char smem[SMEM_BYTES];
  const int tid  = threadIdx.x;
  const int wave = tid >> 5;
  const int lane = tid & 31;
  const int nb   = blockIdx.x;
  const int nn   = lane & 15;
  const int mo   = (lane & 16) ? 8 : 0;
  // Normalized temporal base index (one mod, then add-and-wrap).
  const int tb   = ((tqp[0] % TNUM) + TNUM) % TNUM;

  __builtin_prefetch(W1, 0, 1);
  __builtin_prefetch(W2, 0, 1);
  __builtin_prefetch(Wo, 0, 1);

  _Float16*     q16  = (_Float16*)(smem + G1_OFF);              // [48][80]h
  _Float16*     k16  = q16 + 48 * 80;
  _Float16*     v16  = k16 + 48 * 80;
  unsigned int* wTp  = (unsigned int*)(smem + G1_OFF + 23040);  // [3][16][16]w
  _Float16*     qp   = (_Float16*)(smem + G2_OFF);              // [48][80]h
  unsigned int* kpTp = (unsigned int*)(smem + G2_OFF + 7680);   // [4][16][48]w
  _Float16*     kpTh = (_Float16*)kpTp;
  unsigned int* vpp  = (unsigned int*)(smem + G5_OFF);          // [4][32][16]w
  _Float16*     vph  = (_Float16*)vpp;
  float*        sc   = (float*)(smem + G4_OFF);                 // [4][16][48]f
  _Float16*     attn = (_Float16*)(smem + G1_OFF);              // [4][48][64]h
  _Float16*     outp = (_Float16*)(smem + G2_OFF);              // [48][64]h
  unsigned int* woTp = (unsigned int*)(smem + G2_OFF + 6144);   // [32][64]w
  _Float16*     x16  = (_Float16*)(smem + G1_OFF);              // [48][64]h
  float*        xres = (float*)(smem + G1_OFF + 48 * 64 * 2);   // [48][64]f
  unsigned int* w1Tp = (unsigned int*)(smem + G4_OFF);          // [32][128]w
  unsigned int* w2Tp = (unsigned int*)(smem + G5_OFF);          // [64][64]w
  _Float16*     h16  = (_Float16*)(smem + G2_OFF);              // [48][128]h
  float*        ybuf = (float*)(smem + G4_OFF);                 // [48][64]f

  for (int i = tid; i < SMEM_BYTES / 4; i += 128) ((unsigned int*)smem)[i] = 0u;
  __syncthreads();

  // ---- Phase A: temb add, pack-to-f16 staging of q/k/v; paired W^T tiles --
  {
    unsigned int* qw = (unsigned int*)q16;  // word ld = 40
    unsigned int* kw = (unsigned int*)k16;
    unsigned int* vw = (unsigned int*)v16;
    for (int idx = tid; idx < T * EE / 2; idx += 128) {
      int t = idx >> 5, ew = idx & 31, e = ew << 1;
      int ti = tb + t; if (ti >= TNUM) ti -= TNUM;
      float t0 = temb[ti * EE + e], t1 = temb[ti * EE + e + 1];
      size_t g = (size_t)nb * (T * EE) + t * EE + e;
      qw[t * 40 + ew] = packh2(query[g] + t0, query[g + 1] + t1);
      kw[t * 40 + ew] = packh2(keys[g] + t0,  keys[g + 1] + t1);
      vw[t * 40 + ew] = packh2(value[g], value[g + 1]);
    }
    for (int idx = tid; idx < 3 * 128; idx += 128) {  // paired W^T (rows 8..15 = 0)
      int p = idx >> 7, r = idx & 127, kp = r >> 4, e = r & 15;
      const float* W = (p == 0) ? Wq : ((p == 1) ? Wk : Wv);
      wTp[p * 256 + kp * 16 + e] = packh2(W[e * 16 + 2 * kp], W[e * 16 + 2 * kp + 1]);
    }
  }
  __syncthreads();

  // ---- Phase B: per-head projections (wave == head), K=16 padded to 32 ----
  {
    const int h = wave;
    v16h bq = ldsBp(wTp + 0 * 256, 16);
    v16h bk = ldsBp(wTp + 1 * 256, 16);
    v16h bv = ldsBp(wTp + 2 * 256, 16);
#pragma unroll
    for (int mt = 0; mt < 3; ++mt) {
      const int M0 = mt * 16 + mo;  // even
      v16h aq = ldsA(q16 + mt * 16 * 80 + h * 16, 80);
      v16h ak = ldsA(k16 + mt * 16 * 80 + h * 16, 80);
      v16h av = ldsA(v16 + mt * 16 * 80 + h * 16, 80);
      acch cq = acch_zero(); cq = wmma16h(aq, bq, cq);
      acch ck = acch_zero(); ck = wmma16h(ak, bk, ck);
      acch cv = acch_zero(); cv = wmma16h(av, bv, cv);
      // Base pointers; per-i offsets are compile-time constants.
      _Float16* qdst = qp + M0 * 80 + h * 16 + nn;
      _Float16* kdst = kpTh + h * 1536 + (((nn >> 1) * 48 + M0) << 1) + (nn & 1);
      _Float16* vdst = vph + h * 1024 + (((M0 >> 1) * 16 + nn) << 1);
#pragma unroll
      for (int i = 0; i < 8; ++i) {
        qdst[i * 80]                = acch_get(cq, i);
        kdst[i * 2]                 = acch_get(ck, i);
        vdst[(i >> 1) * 32 + (i & 1)] = acch_get(cv, i);
      }
    }
  }
  __syncthreads();

  // ---- Phase C: energy + mask/scale/clamp/exp in-register, row softmax ----
  {
    const int h = wave;
    v16h b0 = ldsBp(kpTp + h * 768 + 0,  48);
    v16h b1f = ldsBp(kpTp + h * 768 + 16, 48);
    v16h b2f = ldsBp(kpTp + h * 768 + 32, 48);
#pragma unroll
    for (int mt = 0; mt < 3; ++mt) {
      const int M0 = mt * 16 + mo;
      v16h a = ldsA(qp + mt * 16 * 80 + h * 16, 80);
#pragma unroll
      for (int ct = 0; ct < 3; ++ct) {
        v8f c = {};
        c = wmma16(a, (ct == 0) ? b0 : ((ct == 1) ? b1f : b2f), c);
        const int d = ct * 16 + nn - M0;   // masked when kk > m  <=>  d > i
        float* sdst = sc + (h * 16 + mo) * 48 + ct * 16 + nn;
#pragma unroll
        for (int i = 0; i < 8; ++i) {
          float en = (d > i) ? -1e6f : c[i];
          float l = en * 0.125f;                  // 1/sqrt(64)
          l = fminf(5.f, fmaxf(-5.f, l));
          sdst[i * 48] = __expf(l);
        }
      }
      if (lane < 16) {                            // one lane per strip row
        const float4* scr = (const float4*)(sc + (h * 16 + lane) * 48);
        float s = 0.f;
#pragma unroll
        for (int q4 = 0; q4 < 12; ++q4) {
          float4 v4 = scr[q4];
          s += v4.x + v4.y + v4.z + v4.w;
        }
        float inv = 1.f / s;
        uint2* arow = (uint2*)(attn + h * 48 * 64 + (mt * 16 + lane) * 64);
#pragma unroll
        for (int q4 = 0; q4 < 12; ++q4) {
          float4 v4 = scr[q4];
          uint2 w;
          w.x = packh2(v4.x * inv, v4.y * inv);
          w.y = packh2(v4.z * inv, v4.w * inv);
          arow[q4] = w;
        }
        uint2 z; z.x = 0u; z.y = 0u;
#pragma unroll
        for (int q4 = 12; q4 < 16; ++q4) arow[q4] = z;  // zero-pad K 48..63
      }
    }
  }
  __syncthreads();

  // ---- Phase D: attn @ V (K=64 in two chunks); stage paired Wo^T ----
  {
    const int h = wave;
    v16h bv0 = ldsBp(vpp + h * 512 + 0 * 16 * 16, 16);
    v16h bv1 = ldsBp(vpp + h * 512 + 1 * 16 * 16, 16);
#pragma unroll
    for (int mt = 0; mt < 3; ++mt) {
      acch c = acch_zero();
      c = wmma16h(ldsA(attn + h * 48 * 64 + mt * 16 * 64 + 0,  64), bv0, c);
      c = wmma16h(ldsA(attn + h * 48 * 64 + mt * 16 * 64 + 32, 64), bv1, c);
      _Float16* odst = outp + (mt * 16 + mo) * 64 + h * 16 + nn;
#pragma unroll
      for (int i = 0; i < 8; ++i) odst[i * 64] = acch_get(c, i);
    }
  }
  for (int idx = tid; idx < 2048; idx += 128) {   // paired Wo^T
    int kp = idx >> 6, e = idx & 63;
    woTp[idx] = packh2(Wo[e * 64 + 2 * kp], Wo[e * 64 + 2 * kp + 1]);
  }
  __syncthreads();

  // ---- Phase E: out @ Wo^T + bo + residual(q), then LN1 (ct = wave) ----
  {
    const int ct = wave;
    v16h bw0 = ldsBp(woTp + 0 * 16 * 64 + ct * 16, 64);
    v16h bw1 = ldsBp(woTp + 1 * 16 * 64 + ct * 16, 64);
    const int col = ct * 16 + nn;
    const float bov = bo[col];
#pragma unroll
    for (int mt = 0; mt < 3; ++mt) {
      const int M0 = mt * 16 + mo;
      v8f c = {};
      c = wmma16(ldsA(outp + mt * 16 * 64 + 0,  64), bw0, c);
      c = wmma16(ldsA(outp + mt * 16 * 64 + 32, 64), bw1, c);
      const float* qsrc = query + (size_t)nb * (T * EE) + M0 * 64 + col;
      int ti0 = tb + M0;                    // wraps at most once across i
      float* xdst = xres + M0 * 64 + col;
#pragma unroll
      for (int i = 0; i < 8; ++i) {
        int ti = ti0 + i; if (ti >= TNUM) ti -= TNUM;
        xdst[i * 64] = c[i] + bov + qsrc[i * 64] + temb[ti * 64 + col];
      }
    }
  }
  __syncthreads();
  if (tid < T) {
    float4* xr = (float4*)(xres + tid * EE);
    float mu = 0.f, sq = 0.f;
#pragma unroll
    for (int q4 = 0; q4 < 16; ++q4) {
      float4 v4 = xr[q4];
      mu += v4.x + v4.y + v4.z + v4.w;
      sq += v4.x * v4.x + v4.y * v4.y + v4.z * v4.z + v4.w * v4.w;
    }
    mu *= (1.f / EE);
    float var = sq * (1.f / EE) - mu * mu;
    float inv = rsqrtf(var + LNEPS);
    uint2* xh = (uint2*)(x16 + tid * EE);
    const float4* g4 = (const float4*)ln1g;
    const float4* b4 = (const float4*)ln1b;
#pragma unroll
    for (int q4 = 0; q4 < 16; ++q4) {
      float4 v4 = xr[q4], gg = g4[q4], bb = b4[q4];
      float n0 = (v4.x - mu) * inv * gg.x + bb.x;
      float n1 = (v4.y - mu) * inv * gg.y + bb.y;
      float n2 = (v4.z - mu) * inv * gg.z + bb.z;
      float n3 = (v4.w - mu) * inv * gg.w + bb.w;
      uint2 w; w.x = packh2(n0, n1); w.y = packh2(n2, n3);
      xh[q4] = w;                                  // matmul operand (f16)
      xr[q4] = make_float4(n0, n1, n2, n3);        // residual for LN2
    }
  }
  __syncthreads();

  // ---- Phase F: FFN, K streamed in two 128-wide halves, regs accumulate ---
  v8f facc[3];
  facc[0] = (v8f){}; facc[1] = (v8f){}; facc[2] = (v8f){};
  for (int half = 0; half < 2; ++half) {
    for (int idx = tid; idx < 32 * 128; idx += 128) {  // paired W1^T
      int kp = idx >> 7, jj = idx & 127;
      const float* src = W1 + (size_t)(half * 128 + jj) * 64 + 2 * kp;
      w1Tp[idx] = packh2(src[0], src[1]);
    }
    for (int idx = tid; idx < 64 * 64; idx += 128) {   // paired W2^T
      int kp = idx >> 6, e = idx & 63;
      const float* src = W2 + (size_t)e * 256 + half * 128 + 2 * kp;
      w2Tp[idx] = packh2(src[0], src[1]);
    }
    __syncthreads();
    // h1 = relu(x @ W1^T + b1); wave does ct {w, w+4}
#pragma unroll
    for (int mt = 0; mt < 3; ++mt) {
      const int M0 = mt * 16 + mo;
      v16h a0 = ldsA(x16 + mt * 16 * 64 + 0,  64);
      v16h a1 = ldsA(x16 + mt * 16 * 64 + 32, 64);
#pragma unroll
      for (int jj = 0; jj < 2; ++jj) {
        int ct = wave + jj * 4;
        acch c = acch_zero();
        c = wmma16h(a0, ldsBp(w1Tp + 0 * 16 * 128 + ct * 16, 128), c);
        c = wmma16h(a1, ldsBp(w1Tp + 1 * 16 * 128 + ct * 16, 128), c);
        int jl = ct * 16 + nn;
        _Float16 bbh = (_Float16)b1[half * 128 + jl];
        const _Float16 zh = (_Float16)0.f;
        _Float16* hdst = h16 + M0 * 128 + jl;
#pragma unroll
        for (int i = 0; i < 8; ++i) {
          _Float16 t = acch_get(c, i) + bbh;
          hdst[i * 128] = (t > zh) ? t : zh;       // ReLU in f16
        }
      }
    }
    __syncthreads();
    // ff += h1 @ W2^T (K=128 -> 4 chunks); ct = wave, B frags hoisted
    {
      const int ct = wave;
      v16h wb[4];
#pragma unroll
      for (int kc = 0; kc < 4; ++kc)
        wb[kc] = ldsBp(w2Tp + kc * 16 * 64 + ct * 16, 64);
#pragma unroll
      for (int mt = 0; mt < 3; ++mt) {
#pragma unroll
        for (int kc = 0; kc < 4; ++kc)
          facc[mt] = wmma16(ldsA(h16 + mt * 16 * 128 + kc * 32, 128), wb[kc],
                            facc[mt]);
      }
    }
    __syncthreads();
  }

  // ---- Phase G: + b2 + residual, LN2, store ----
  {
    const int col = wave * 16 + nn;
    const float b2v = b2[col];
#pragma unroll
    for (int mt = 0; mt < 3; ++mt) {
      const int M0 = mt * 16 + mo;
      float*       ydst = ybuf + M0 * 64 + col;
      const float* xsrc = xres + M0 * 64 + col;
#pragma unroll
      for (int i = 0; i < 8; ++i)
        ydst[i * 64] = facc[mt][i] + b2v + xsrc[i * 64];
    }
  }
  __syncthreads();
  if (tid < T) {
    const float4* yr = (const float4*)(ybuf + tid * EE);
    float mu = 0.f, sq = 0.f;
#pragma unroll
    for (int q4 = 0; q4 < 16; ++q4) {
      float4 v4 = yr[q4];
      mu += v4.x + v4.y + v4.z + v4.w;
      sq += v4.x * v4.x + v4.y * v4.y + v4.z * v4.z + v4.w * v4.w;
    }
    mu *= (1.f / EE);
    float var = sq * (1.f / EE) - mu * mu;
    float inv = rsqrtf(var + LNEPS);
    float4* yo = (float4*)(out + (size_t)nb * (T * EE) + tid * EE);
    const float4* g4 = (const float4*)ln2g;
    const float4* b4 = (const float4*)ln2b;
#pragma unroll
    for (int q4 = 0; q4 < 16; ++q4) {
      float4 v4 = yr[q4], gg = g4[q4], bb = b4[q4];
      yo[q4] = make_float4((v4.x - mu) * inv * gg.x + bb.x,
                           (v4.y - mu) * inv * gg.y + bb.y,
                           (v4.z - mu) * inv * gg.z + bb.z,
                           (v4.w - mu) * inv * gg.w + bb.w);
    }
  }
}

extern "C" void kernel_launch(void* const* d_in, const int* in_sizes, int n_in,
                              void* d_out, int out_size, void* d_ws, size_t ws_size,
                              hipStream_t stream) {
  (void)in_sizes; (void)n_in; (void)d_ws; (void)ws_size; (void)out_size;
  const float* value = (const float*)d_in[0];
  const float* keys  = (const float*)d_in[1];
  const float* query = (const float*)d_in[2];
  const float* Wv    = (const float*)d_in[3];
  const float* Wk    = (const float*)d_in[4];
  const float* Wq    = (const float*)d_in[5];
  const float* Wo    = (const float*)d_in[6];
  const float* bo    = (const float*)d_in[7];
  const float* temb  = (const float*)d_in[8];
  const float* ln1g  = (const float*)d_in[9];
  const float* ln1b  = (const float*)d_in[10];
  const float* ln2g  = (const float*)d_in[11];
  const float* ln2b  = (const float*)d_in[12];
  const float* W1    = (const float*)d_in[13];
  const float* b1    = (const float*)d_in[14];
  const float* W2    = (const float*)d_in[15];
  const float* b2    = (const float*)d_in[16];
  const int*   tq    = (const int*)d_in[17];
  float* out = (float*)d_out;

  ttransformer_fused<<<NB, 128, 0, stream>>>(value, keys, query, Wq, Wk, Wv, Wo,
                                             bo, temb, ln1g, ln1b, ln2g, ln2b,
                                             W1, b1, W2, b2, tq, out);
}
